// MultiHeadAttention_59511066853520
// MI455X (gfx1250) — matrix operand order repaired and checked by reading, not compiled, not run on hardware
//
#include <hip/hip_runtime.h>
#include <hip/hip_bf16.h>
#include <math.h>

// ---------------------------------------------------------------------------
// MultiHeadAttention with relative-position bias, MI455X (gfx1250, wave32)
//   - bias folded into K:  K'' = log2e * (scale*K + R),  softmax in log2 domain
//   - all GEMMs / attention via v_wmma_f32_16x16x32_bf16 (fp32 accumulate)
//   - flash-style online softmax, 64-key chunks (8 WMMAs per softmax update)
//   - all fragment traffic as b128 loads (weights pre-transposed to bf16,
//     V stored transposed so every WMMA operand is contiguous per lane)
// ---------------------------------------------------------------------------

#define BATCH 4
#define SEQ   2048
#define DMODEL 256
#define NHEAD 8
#define HDIM  32
#define MAXREL 32
#define LOG2E 1.4426950408889634f

typedef __attribute__((ext_vector_type(16))) __bf16 v16bf;
typedef __attribute__((ext_vector_type(8)))  float  v8f;

union bfrag_t { v16bf v; uint4 q[2]; __bf16 e[16]; };

// A-fragment (16x32 bf16) per-lane element -> K index as loaded here (same k order as B):
//   e in [0,8)  -> K = 16*hi + e       (contiguous run 0)
//   e in [8,16) -> K = 16*hi + e       (contiguous run 1)
// B-fragment (32x16 bf16): e -> K = 16*hi + e (one contiguous 32B run)

// ---------------------------------------------------------------------------
// Kernel 0: R[j][d] = sum_k table[clip(j-k,-32,32)+32][d]
// ---------------------------------------------------------------------------
__global__ void relsum_kernel(const float* __restrict__ table, float* __restrict__ R) {
    int t = blockIdx.x * blockDim.x + threadIdx.x;   // 0 .. SEQ*HDIM-1
    int j = t >> 5, d = t & 31;
    float sum = 0.f;
    int c_lo = SEQ - MAXREL - j; if (c_lo < 0) c_lo = 0;   // bucket 0  (rel <= -32)
    int c_hi = j - (MAXREL - 1); if (c_hi < 0) c_hi = 0;   // bucket 64 (rel >= 32)
    sum += (float)c_lo * table[0 * HDIM + d];
    sum += (float)c_hi * table[(2 * MAXREL) * HDIM + d];
    #pragma unroll 8
    for (int rel = -(MAXREL - 1); rel <= (MAXREL - 1); ++rel) {
        int k = j - rel;
        if (k >= 0 && k < SEQ) sum += table[(rel + MAXREL) * HDIM + d];
    }
    R[j * HDIM + d] = sum;
}

// ---------------------------------------------------------------------------
// Prep: query [8192,256] f32 -> bf16 (vectorized)
// ---------------------------------------------------------------------------
__global__ void cvt_x_kernel(const float* __restrict__ X, __bf16* __restrict__ Xb) {
    int i = blockIdx.x * blockDim.x + threadIdx.x;   // one float4 each
    float4 f = ((const float4*)X)[i];
    union { uint2 u; __bf16 e[4]; } p;
    p.e[0] = (__bf16)f.x; p.e[1] = (__bf16)f.y;
    p.e[2] = (__bf16)f.z; p.e[3] = (__bf16)f.w;
    ((uint2*)Xb)[i] = p.u;
}

// ---------------------------------------------------------------------------
// Prep: weights f32 [k][n] -> bf16 transposed Wt[n][k]  (z picks matrix)
// ---------------------------------------------------------------------------
__global__ void cvt_w_kernel(const float* __restrict__ Wq, const float* __restrict__ Wk,
                             const float* __restrict__ Wv, const float* __restrict__ Wo,
                             __bf16* __restrict__ Wt) {
    const float* W = (blockIdx.z == 0) ? Wq : (blockIdx.z == 1) ? Wk
                   : (blockIdx.z == 2) ? Wv : Wo;
    __bf16* dst = Wt + (size_t)blockIdx.z * DMODEL * DMODEL;
    int t = blockIdx.x * blockDim.x + threadIdx.x;   // 0..65535
    int n = t >> 8, k = t & 255;
    dst[t] = (__bf16)W[k * DMODEL + n];              // coalesced writes
}

// ---------------------------------------------------------------------------
// Kernel 1: fused QKV projection.  blockIdx.z: 0=Q, 1=K''(log2-scaled+R), 2=V.
// One 16x16 output tile per wave; 8 WMMAs over K=256; all operands b128.
// Q,K layout [B,H,S,32]; V layout transposed [B,H,32,S].
// ---------------------------------------------------------------------------
__global__ void __launch_bounds__(128)
proj_kernel(const __bf16* __restrict__ Xb, const __bf16* __restrict__ Wt,
            const float* __restrict__ bq, const float* __restrict__ bk,
            const float* __restrict__ bv, const float* __restrict__ R,
            __bf16* __restrict__ Qb, __bf16* __restrict__ Kb, __bf16* __restrict__ Vt)
{
    const int mode = blockIdx.z;
    const __bf16* Wm = Wt + (size_t)mode * DMODEL * DMODEL;
    const float* bias = (mode == 0) ? bq : (mode == 1) ? bk : bv;

    const int lane = threadIdx.x & 31;
    const int wave = threadIdx.x >> 5;
    const int tile = blockIdx.x * 4 + wave;          // 0..8191
    const int tM = tile >> 4, tN = tile & 15;        // 512 x 16 tiles
    const int ln = lane & 15, hi = lane >> 4;

    const __bf16* arow = Xb + (size_t)(tM * 16 + ln) * DMODEL;
    const __bf16* brow = Wm + (size_t)(tN * 16 + ln) * DMODEL;

    v8f acc = {};
    #pragma unroll
    for (int kb = 0; kb < DMODEL; kb += 32) {
        bfrag_t a, bf;
        a.q[0] = *(const uint4*)(arow + kb + 16 * hi);
        a.q[1] = *(const uint4*)(arow + kb + 16 * hi + 8);
        const uint4* bp = (const uint4*)(brow + kb + 16 * hi);
        bf.q[0] = bp[0];
        bf.q[1] = bp[1];
        acc = __builtin_amdgcn_wmma_f32_16x16x32_bf16(
                  false, a.v, false, bf.v, (short)0, acc, false, false);
    }

    const int col = tN * 16 + ln;                    // 0..255
    const int h = col >> 5, d = col & 31;
    const float bval = bias[col];
    // mode 1: K'' = log2e * (scale*(xW+b) + R)  -> softmax done with exp2
    const float sfac = (mode == 1) ? (0.17677669529663687f * LOG2E) : 1.0f;
    const float rmul = (mode == 1) ? LOG2E : 0.0f;

    const int row0 = tM * 16 + 8 * hi;               // first token of this half
    const int b = row0 >> 11, s0 = row0 & (SEQ - 1);

    if (mode != 2) {
        __bf16* Out = (mode == 0) ? Qb : Kb;
        #pragma unroll
        for (int r = 0; r < 8; ++r) {
            int s = s0 + r;
            float v = (acc[r] + bval) * sfac + rmul * R[s * HDIM + d];
            Out[(((size_t)(b * NHEAD + h)) * SEQ + s) * HDIM + d] = (__bf16)v;
        }
    } else {
        union { uint4 q; __bf16 e[8]; } pk;
        #pragma unroll
        for (int r = 0; r < 8; ++r) pk.e[r] = (__bf16)(acc[r] + bval);
        // transposed: Vt[((b*H+h)*32 + d)*SEQ + s], rows r contiguous in s
        *(uint4*)(Vt + (((size_t)(b * NHEAD + h)) * HDIM + d) * SEQ + s0) = pk.q;
    }
}

// ---------------------------------------------------------------------------
// Kernel 2: flash attention.  One wave = 16 query rows of one (b,h).
// 64-key chunks: 4 score WMMAs + one softmax update + 4 PV WMMAs.
// ---------------------------------------------------------------------------
__global__ void __launch_bounds__(128)
attn_kernel(const __bf16* __restrict__ Qb, const __bf16* __restrict__ Kb,
            const __bf16* __restrict__ Vt, const float* __restrict__ mask,
            __bf16* __restrict__ AO)
{
    __shared__ __align__(16) float P[4][16][64];     // per-wave P tile (16 KiB)

    const int lane = threadIdx.x & 31;
    const int w    = threadIdx.x >> 5;
    const int id   = blockIdx.x * 4 + w;             // 0..4095
    const int qt = id & 127;
    const int h  = (id >> 7) & 7;
    const int b  = id >> 10;
    const int ln = lane & 15, hi = lane >> 4;

    const __bf16* Qh = Qb + ((size_t)(b * NHEAD + h)) * SEQ * HDIM;
    const __bf16* Kh = Kb + ((size_t)(b * NHEAD + h)) * SEQ * HDIM;
    const __bf16* Vh = Vt + ((size_t)(b * NHEAD + h)) * HDIM * SEQ;  // [32][2048]
    const float*  mk = mask + (size_t)b * SEQ;

    // Q A-fragment (two 16B runs per lane), resident for the whole loop
    bfrag_t qf;
    {
        const __bf16* qrow = Qh + (size_t)(qt * 16 + ln) * HDIM;
        qf.q[0] = *(const uint4*)(qrow + 16 * hi);
        qf.q[1] = *(const uint4*)(qrow + 16 * hi + 8);
    }

    v8f o0 = {}, o1 = {};
    float mrow[8], lrow[8];
    #pragma unroll
    for (int r = 0; r < 8; ++r) { mrow[r] = -3.0e38f; lrow[r] = 0.0f; }

    const __bf16* v0row = Vh + (size_t)ln * SEQ;         // d = ln
    const __bf16* v1row = Vh + (size_t)(16 + ln) * SEQ;  // d = 16+ln

    for (int jc = 0; jc < SEQ; jc += 64) {
        // K'' B-fragments: one contiguous 32B run per lane each (4 x 16 keys)
        v8f s[4];
        {
            bfrag_t k0, k1, k2, k3;
            const size_t kb0 = (size_t)(jc + ln) * HDIM + 16 * hi;
            k0.q[0] = *(const uint4*)(Kh + kb0);
            k0.q[1] = *(const uint4*)(Kh + kb0 + 8);
            k1.q[0] = *(const uint4*)(Kh + kb0 + 16 * HDIM);
            k1.q[1] = *(const uint4*)(Kh + kb0 + 16 * HDIM + 8);
            k2.q[0] = *(const uint4*)(Kh + kb0 + 32 * HDIM);
            k2.q[1] = *(const uint4*)(Kh + kb0 + 32 * HDIM + 8);
            k3.q[0] = *(const uint4*)(Kh + kb0 + 48 * HDIM);
            k3.q[1] = *(const uint4*)(Kh + kb0 + 48 * HDIM + 8);
            v8f z = {};
            s[0] = __builtin_amdgcn_wmma_f32_16x16x32_bf16(
                       false, qf.v, false, k0.v, (short)0, z, false, false);
            s[1] = __builtin_amdgcn_wmma_f32_16x16x32_bf16(
                       false, qf.v, false, k1.v, (short)0, z, false, false);
            s[2] = __builtin_amdgcn_wmma_f32_16x16x32_bf16(
                       false, qf.v, false, k2.v, (short)0, z, false, false);
            s[3] = __builtin_amdgcn_wmma_f32_16x16x32_bf16(
                       false, qf.v, false, k3.v, (short)0, z, false, false);
        }

        // mask additive term, pre-scaled to log2 domain
        float am[4];
        #pragma unroll
        for (int c = 0; c < 4; ++c)
            am[c] = (1.0f - mk[jc + 16 * c + ln]) * (-1.0e9f * LOG2E);

        // online softmax update for 64 keys at once (base-2 exponentials)
        #pragma unroll
        for (int r = 0; r < 8; ++r) {
            float x0 = s[0][r] + am[0];
            float x1 = s[1][r] + am[1];
            float x2 = s[2][r] + am[2];
            float x3 = s[3][r] + am[3];
            float cm = fmaxf(fmaxf(x0, x1), fmaxf(x2, x3));
            #pragma unroll
            for (int off = 1; off < 16; off <<= 1)
                cm = fmaxf(cm, __shfl_xor(cm, off, 32));
            float mn    = fmaxf(mrow[r], cm);
            float alpha = exp2f(mrow[r] - mn);
            float p0 = exp2f(x0 - mn);
            float p1 = exp2f(x1 - mn);
            float p2 = exp2f(x2 - mn);
            float p3 = exp2f(x3 - mn);
            float rs = (p0 + p1) + (p2 + p3);
            #pragma unroll
            for (int off = 1; off < 16; off <<= 1)
                rs += __shfl_xor(rs, off, 32);
            lrow[r] = lrow[r] * alpha + rs;
            mrow[r] = mn;
            o0[r] *= alpha;  o1[r] *= alpha;
            P[w][r + 8 * hi][ln]      = p0;
            P[w][r + 8 * hi][ln + 16] = p1;
            P[w][r + 8 * hi][ln + 32] = p2;
            P[w][r + 8 * hi][ln + 48] = p3;
        }
        asm volatile("s_wait_dscnt 0" ::: "memory");

        // P as two A-fragments (keys 0..31 and 32..63): float4 LDS reads
        v16bf pa0, pa1;
        {
            float pf[16];
            *(float4*)&pf[0]  = *(const float4*)&P[w][ln][16 * hi];
            *(float4*)&pf[4]  = *(const float4*)&P[w][ln][16 * hi + 4];
            *(float4*)&pf[8]  = *(const float4*)&P[w][ln][16 * hi + 8];
            *(float4*)&pf[12] = *(const float4*)&P[w][ln][16 * hi + 12];
            #pragma unroll
            for (int e = 0; e < 16; ++e) pa0[e] = (__bf16)pf[e];
            *(float4*)&pf[0]  = *(const float4*)&P[w][ln][32 + 16 * hi];
            *(float4*)&pf[4]  = *(const float4*)&P[w][ln][32 + 16 * hi + 4];
            *(float4*)&pf[8]  = *(const float4*)&P[w][ln][32 + 16 * hi + 8];
            *(float4*)&pf[12] = *(const float4*)&P[w][ln][32 + 16 * hi + 12];
            #pragma unroll
            for (int e = 0; e < 16; ++e) pa1[e] = (__bf16)pf[e];
        }

        // V B-fragments from transposed Vt: contiguous 32B runs per lane
        {
            bfrag_t v0a, v1a, v0b, v1b;
            const uint4* vp0 = (const uint4*)(v0row + jc + 16 * hi);
            const uint4* vp1 = (const uint4*)(v1row + jc + 16 * hi);
            const uint4* vq0 = (const uint4*)(v0row + jc + 32 + 16 * hi);
            const uint4* vq1 = (const uint4*)(v1row + jc + 32 + 16 * hi);
            v0a.q[0] = vp0[0]; v0a.q[1] = vp0[1];
            v1a.q[0] = vp1[0]; v1a.q[1] = vp1[1];
            v0b.q[0] = vq0[0]; v0b.q[1] = vq0[1];
            v1b.q[0] = vq1[0]; v1b.q[1] = vq1[1];
            o0 = __builtin_amdgcn_wmma_f32_16x16x32_bf16(
                     false, pa0, false, v0a.v, (short)0, o0, false, false);
            o1 = __builtin_amdgcn_wmma_f32_16x16x32_bf16(
                     false, pa0, false, v1a.v, (short)0, o1, false, false);
            o0 = __builtin_amdgcn_wmma_f32_16x16x32_bf16(
                     false, pa1, false, v0b.v, (short)0, o0, false, false);
            o1 = __builtin_amdgcn_wmma_f32_16x16x32_bf16(
                     false, pa1, false, v1b.v, (short)0, o1, false, false);
        }
    }

    #pragma unroll
    for (int r = 0; r < 8; ++r) {
        int s = qt * 16 + r + 8 * hi;
        float inv = 1.0f / lrow[r];
        size_t base = ((size_t)(b * SEQ + s)) * DMODEL + h * HDIM;
        AO[base + ln]      = (__bf16)(o0[r] * inv);
        AO[base + 16 + ln] = (__bf16)(o1[r] * inv);
    }
}

// ---------------------------------------------------------------------------
// Kernel 3: output projection  AO[8192,256](bf16) @ Wo^T(bf16) + bo -> f32
// ---------------------------------------------------------------------------
__global__ void __launch_bounds__(128)
oproj_kernel(const __bf16* __restrict__ AO, const __bf16* __restrict__ Wto,
             const float* __restrict__ bo, float* __restrict__ out)
{
    const int lane = threadIdx.x & 31;
    const int wave = threadIdx.x >> 5;
    const int tile = blockIdx.x * 4 + wave;
    const int tM = tile >> 4, tN = tile & 15;
    const int ln = lane & 15, hi = lane >> 4;

    const __bf16* arow = AO  + (size_t)(tM * 16 + ln) * DMODEL;
    const __bf16* brow = Wto + (size_t)(tN * 16 + ln) * DMODEL;

    v8f acc = {};
    #pragma unroll
    for (int kb = 0; kb < DMODEL; kb += 32) {
        bfrag_t a, bf;
        a.q[0] = *(const uint4*)(arow + kb + 16 * hi);
        a.q[1] = *(const uint4*)(arow + kb + 16 * hi + 8);
        const uint4* bp = (const uint4*)(brow + kb + 16 * hi);
        bf.q[0] = bp[0];
        bf.q[1] = bp[1];
        acc = __builtin_amdgcn_wmma_f32_16x16x32_bf16(
                  false, a.v, false, bf.v, (short)0, acc, false, false);
    }
    const float bval = bo[tN * 16 + ln];
    #pragma unroll
    for (int r = 0; r < 8; ++r)
        out[(size_t)(tM * 16 + r + 8 * hi) * DMODEL + tN * 16 + ln] = acc[r] + bval;
}

// ---------------------------------------------------------------------------
extern "C" void kernel_launch(void* const* d_in, const int* in_sizes, int n_in,
                              void* d_out, int out_size, void* d_ws, size_t ws_size,
                              hipStream_t stream) {
    const float* query = (const float*)d_in[0];
    const float* mask  = (const float*)d_in[1];
    const float* Wq    = (const float*)d_in[2];
    const float* bq    = (const float*)d_in[3];
    const float* Wk    = (const float*)d_in[4];
    const float* bk    = (const float*)d_in[5];
    const float* Wv    = (const float*)d_in[6];
    const float* bv    = (const float*)d_in[7];
    const float* Wo    = (const float*)d_in[8];
    const float* bo    = (const float*)d_in[9];
    const float* table = (const float*)d_in[10];

    // Workspace layout (all regions 256B+ aligned)
    char* ws = (char*)d_ws;
    float*  R  = (float*)ws;                                  // 2048*32 f32   (256 KiB)
    __bf16* Xb = (__bf16*)(ws + (size_t)SEQ * HDIM * 4);      // [8192,256]    (4 MiB)
    __bf16* Wt = Xb + (size_t)BATCH * SEQ * DMODEL;           // 4x[256,256]   (512 KiB)
    __bf16* Qb = Wt + (size_t)4 * DMODEL * DMODEL;            // [B,H,S,32]    (4 MiB)
    const size_t qkv_elems = (size_t)BATCH * NHEAD * SEQ * HDIM;
    __bf16* Kb = Qb + qkv_elems;                              // [B,H,S,32]
    __bf16* Vt = Kb + qkv_elems;                              // [B,H,32,S]
    __bf16* AO = Vt + qkv_elems;                              // [B,S,256]

    // 0) relative-position row sums R[S,32]
    relsum_kernel<<<(SEQ * HDIM) / 256, 256, 0, stream>>>(table, R);

    // 0b) one-time precision/layout prep
    cvt_x_kernel<<<(BATCH * SEQ * DMODEL / 4) / 256, 256, 0, stream>>>(query, Xb);
    cvt_w_kernel<<<dim3(DMODEL * DMODEL / 256, 1, 4), 256, 0, stream>>>(Wq, Wk, Wv, Wo, Wt);

    // 1) Q / K''(log2-domain, rel-bias folded) / V projections
    proj_kernel<<<dim3(2048, 1, 3), 128, 0, stream>>>(
        Xb, Wt, bq, bk, bv, R, Qb, Kb, Vt);

    // 2) flash attention -> AO bf16 [B,S,256]
    attn_kernel<<<1024, 128, 0, stream>>>(Qb, Kb, Vt, mask, AO);

    // 3) output projection -> fp32 d_out [B,S,256]
    oproj_kernel<<<2048, 128, 0, stream>>>(
        AO, Wt + (size_t)3 * DMODEL * DMODEL, bo, (float*)d_out);
}
